// Block_41893111005419
// MI455X (gfx1250) — matrix operand order repaired and checked by reading, not compiled, not run on hardware
//
#include <hip/hip_runtime.h>
#include <hip/hip_bf16.h>

// ---------------------------------------------------------------------------
// Types for CDNA5 WMMA (wave32): bf16 A/B fragments, f32 accumulators.
// ---------------------------------------------------------------------------
typedef __attribute__((ext_vector_type(16))) __bf16 v16bf;
typedef __attribute__((ext_vector_type(8)))  float   v8f;
typedef __attribute__((ext_vector_type(2)))  __bf16 bf16x2;
typedef __attribute__((ext_vector_type(4)))  __bf16 bf16x4;
typedef __attribute__((ext_vector_type(8)))  __bf16 bf16x8;
typedef unsigned int u32x4 __attribute__((ext_vector_type(4)));
typedef int          i32x4 __attribute__((ext_vector_type(4)));
typedef int          i32x8 __attribute__((ext_vector_type(8)));

// Tensor Data Mover availability (device pass only; host pass parses fallback).
#if defined(__HIP_DEVICE_COMPILE__) && __has_builtin(__builtin_amdgcn_tensor_load_to_lds) && \
    __has_builtin(__builtin_amdgcn_s_wait_tensorcnt)
#define USE_TDM 1
#else
#define USE_TDM 0
#endif

__device__ __forceinline__ v8f wmma_bf16(v16bf a, v16bf b, v8f c) {
  // D = A(16x32 bf16) * B(32x16 bf16) + C(16x16 f32)
  return __builtin_amdgcn_wmma_f32_16x16x32_bf16(
      /*neg_a=*/false, a, /*neg_b=*/false, b,
      /*c_mod=*/(short)0, c, /*reuse_a=*/false, /*reuse_b=*/false);
}

__device__ __forceinline__ v8f v8f_zero() {
  v8f z;
#pragma unroll
  for (int e = 0; e < 8; ++e) z[e] = 0.0f;
  return z;
}

__device__ __forceinline__ float gelu_f(float x) {
  float t = x + 0.044715f * x * x * x;
  return 0.5f * x * (1.0f + tanhf(0.7978845608028654f * t));
}

#if USE_TDM
// ---------------------------------------------------------------------------
// TDM 2D tile load: global (row-major, 48-bit stride) -> LDS with row padding.
// Descriptor per CDNA5 ISA:
//   group0: count=1 | lds_addr[63:32] | global_addr[120:64] | type=2[127:126]
//   group1: data_size=2B | pad ctl | tensor_dim0/1 | tile_dim0/1 | dim0_stride
// pad_interval code c -> pad after 2^(c+1) DWORDs; pad_amount code c -> c+1 DWORDs.
// ---------------------------------------------------------------------------
__device__ __forceinline__ void tdm_load_2d_bf16(
    unsigned lds_off, const void* gptr,
    unsigned tensor_d0, unsigned tensor_d1, unsigned long long stride0_elems,
    unsigned tile_d0, unsigned tile_d1,
    unsigned pad_interval_code, unsigned pad_amount_code) {
  unsigned long long ga = (unsigned long long)gptr;
  u32x4 g0;
  g0[0] = 1u;                                               // count=1, user mode
  g0[1] = lds_off;                                          // lds_addr
  g0[2] = (unsigned)ga;                                     // global_addr[31:0]
  g0[3] = (unsigned)((ga >> 32) & 0x1FFFFFFull) | (2u << 30); // addr[56:32] | type=2
  i32x8 g1;
  g1[0] = (int)((1u << 16) |                // data_size = 1 -> 2 bytes
                (1u << 20) |                // pad_enable
                (pad_interval_code << 22) |
                (pad_amount_code << 25));
  g1[1] = (int)((tensor_d0 & 0xFFFFu) << 16);                       // dim0[15:0]
  g1[2] = (int)(((tensor_d0 >> 16) & 0xFFFFu) |
                ((tensor_d1 & 0xFFFFu) << 16));                     // dim0[31:16]|dim1[15:0]
  g1[3] = (int)(((tensor_d1 >> 16) & 0xFFFFu) |
                ((tile_d0 & 0xFFFFu) << 16));                       // dim1[31:16]|tile0
  g1[4] = (int)(tile_d1 & 0xFFFFu);                                 // tile1 | tile2=0
  g1[5] = (int)(unsigned)(stride0_elems & 0xFFFFFFFFull);           // stride0[31:0]
  g1[6] = (int)(unsigned)((stride0_elems >> 32) & 0xFFFFull);       // stride0[47:32]
  g1[7] = 0;
  i32x4 z4 = {0, 0, 0, 0};
#if __has_include(<hip/amd_detail/amd_gfx1250_TDM.h>)
  i32x8 z8 = {0, 0, 0, 0, 0, 0, 0, 0};
  __builtin_amdgcn_tensor_load_to_lds(g0, g1, z4, z4, z8, 0);
#else
  __builtin_amdgcn_tensor_load_to_lds(g0, g1, z4, z4, 0);
#endif
}
#endif  // USE_TDM

// ---------------------------------------------------------------------------
// fp32 -> bf16 conversion (weights), vectorized 4-wide.
// ---------------------------------------------------------------------------
__global__ __launch_bounds__(256) void cvt_kernel(const float* __restrict__ s,
                                                  __bf16* __restrict__ d, int n4) {
  int i = blockIdx.x * 256 + threadIdx.x;
  if (i < n4) {
    float4 v = ((const float4*)s)[i];
    bf16x4 o;
    o.x = (__bf16)v.x; o.y = (__bf16)v.y; o.z = (__bf16)v.z; o.w = (__bf16)v.w;
    ((bf16x4*)d)[i] = o;
  }
}

// ---------------------------------------------------------------------------
// LayerNorm over C=1024, one row per 256-thread block, bf16 output.
// ---------------------------------------------------------------------------
__global__ __launch_bounds__(256) void ln_kernel(const float* __restrict__ xin,
                                                 const float* __restrict__ w,
                                                 const float* __restrict__ bb,
                                                 __bf16* __restrict__ out) {
  const int row = blockIdx.x;
  const int tid = threadIdx.x;
  float4 v = ((const float4*)(xin + (size_t)row * 1024))[tid];
  float s = v.x + v.y + v.z + v.w;
  float q = v.x * v.x + v.y * v.y + v.z * v.z + v.w * v.w;
#pragma unroll
  for (int off = 16; off > 0; off >>= 1) {
    s += __shfl_xor(s, off);
    q += __shfl_xor(q, off);
  }
  __shared__ float rs[8], rq[8];
  const int wid = tid >> 5, lane = tid & 31;
  if (lane == 0) { rs[wid] = s; rq[wid] = q; }
  __syncthreads();
  float ts = 0.f, tq = 0.f;
#pragma unroll
  for (int i = 0; i < 8; ++i) { ts += rs[i]; tq += rq[i]; }
  const float mean = ts * (1.0f / 1024.0f);
  const float var  = tq * (1.0f / 1024.0f) - mean * mean;
  const float rstd = rsqrtf(var + 1e-5f);
  float4 wv = ((const float4*)w)[tid];
  float4 bv = ((const float4*)bb)[tid];
  bf16x4 o;
  o.x = (__bf16)((v.x - mean) * rstd * wv.x + bv.x);
  o.y = (__bf16)((v.y - mean) * rstd * wv.y + bv.y);
  o.z = (__bf16)((v.z - mean) * rstd * wv.z + bv.z);
  o.w = (__bf16)((v.w - mean) * rstd * wv.w + bv.w);
  ((bf16x4*)(out + (size_t)row * 1024))[tid] = o;
}

// ---------------------------------------------------------------------------
// Fragment gathers from LDS (A/B: this file's contiguous k order, same for both; C: ISA layout):
//  A 16x32 bf16: lane l -> row l%16; VGPR v holds K = (l/16)*16 + 2v
//  B 32x16 bf16: lane l -> col l%16; VGPR v holds K = (l/16)*16 + 2v
//  C 16x16 f32 : VGPR e, lane l -> M = e + 8*(l/16), N = l%16
// ---------------------------------------------------------------------------
__device__ __forceinline__ v16bf load_a_frag(const __bf16* As, int row, int stride) {
  const int l = threadIdx.x & 31;
  const int half = l >> 4;
  const __bf16* p = As + (size_t)(row + (l & 15)) * stride;
  v16bf a;
#pragma unroll
  for (int v = 0; v < 8; ++v) {
    int kb = half * 16 + v * 2;
    bf16x2 t = *(const bf16x2*)(p + kb);
    a[2 * v] = t.x; a[2 * v + 1] = t.y;
  }
  return a;
}

__device__ __forceinline__ v16bf load_b_frag(const __bf16* BsT, int col, int stride) {
  const int l = threadIdx.x & 31;
  const int half = l >> 4;
  const __bf16* p = BsT + (size_t)(col + (l & 15)) * stride + half * 16;
  v16bf b;
#pragma unroll
  for (int v = 0; v < 8; ++v) {
    bf16x2 t = *(const bf16x2*)(p + 2 * v);
    b[2 * v] = t.x; b[2 * v + 1] = t.y;
  }
  return b;
}

// ---------------------------------------------------------------------------
// Tiled WMMA GEMM: C[M,N] = A[M,K](bf16) * B[K,N](bf16) + epilogue
// Block tile 128x128, BK=32; 8 waves in 4x2 grid, each wave 32x64 (2x4 WMMA tiles).
// A tile staged by TDM (padded rows); B tile staged transposed by the block.
// EPI 0: out_bf16 = acc + bias
// EPI 1: out_bf16 = gelu(acc + bias)
// EPI 2: out_f32  = acc + bias + res
// ---------------------------------------------------------------------------
#define BM 128
#define BN 128
#define BK 32
#define LDA_S 40  // 32 elems (16 DWORDs) + 8 elems (4 DWORDs) pad
#define LDB_S 40

template <int EPI>
__global__ __launch_bounds__(256) void gemm_kernel(
    const __bf16* __restrict__ A, const __bf16* __restrict__ Bw,
    const float* __restrict__ bias, const float* __restrict__ res,
    float* __restrict__ outF, __bf16* __restrict__ outB, int N, int K) {
  __shared__ __bf16 As[BM * LDA_S];
  __shared__ __bf16 BsT[BN * LDB_S];  // stored transposed: [n][k]

  const int tid = threadIdx.x;
  const int bm = blockIdx.y * BM;
  const int bn = blockIdx.x * BN;
  const int w  = tid >> 5;
  const int wm = (w >> 1) * 32;
  const int wn = (w & 1) * 64;

  v8f c[2][4];
#pragma unroll
  for (int i = 0; i < 2; ++i)
#pragma unroll
    for (int j = 0; j < 4; ++j) c[i][j] = v8f_zero();

  const int steps = K / BK;
  for (int s = 0; s < steps; ++s) {
    const int k0 = s * BK;
    __syncthreads();
#if USE_TDM
    // A tile 128x32 via Tensor Data Mover (wave 0 issues; EXEC-independent,
    // overlaps with the manual B staging below). pad: 16 DWORDs -> +4 DWORDs.
    if (w == 0) {
      tdm_load_2d_bf16((unsigned)(unsigned long long)(uintptr_t)As,
                       A + (size_t)bm * K + k0,
                       /*tensor_d0=*/BK, /*tensor_d1=*/BM,
                       /*stride0=*/(unsigned long long)K,
                       /*tile_d0=*/BK, /*tile_d1=*/BM,
                       /*pad_interval=*/3, /*pad_amount=*/3);
    }
#else
    // A tile: 128x32, two b128 passes per thread (row-major in LDS).
#pragma unroll
    for (int p = 0; p < 2; ++p) {
      int e = (p * 256 + tid) * 8;
      int r = e >> 5, cc = e & 31;
      bf16x8 va = *(const bf16x8*)(A + (size_t)(bm + r) * K + k0 + cc);
      *(bf16x8*)(As + (size_t)r * LDA_S + cc) = va;
    }
#endif
    // B tile: 32x128, b128 loads, transposed scalar stores -> BsT[n][k].
#pragma unroll
    for (int p = 0; p < 2; ++p) {
      int e = (p * 256 + tid) * 8;
      int r = e >> 7, cc = e & 127;
      bf16x8 vb = *(const bf16x8*)(Bw + (size_t)(k0 + r) * N + bn + cc);
#pragma unroll
      for (int q = 0; q < 8; ++q) BsT[(size_t)(cc + q) * LDB_S + r] = vb[q];
    }
#if USE_TDM
    if (w == 0) __builtin_amdgcn_s_wait_tensorcnt(0);
#endif
    __syncthreads();
    v16bf a0 = load_a_frag(As, wm, LDA_S);
    v16bf a1 = load_a_frag(As, wm + 16, LDA_S);
#pragma unroll
    for (int j = 0; j < 4; ++j) {
      v16bf b = load_b_frag(BsT, wn + j * 16, LDB_S);
      c[0][j] = wmma_bf16(a0, b, c[0][j]);
      c[1][j] = wmma_bf16(a1, b, c[1][j]);
    }
  }

  // Epilogue per C-layout element.
  const int l = tid & 31;
  const int col0 = l & 15;
  const int gh = l >> 4;
#pragma unroll
  for (int i = 0; i < 2; ++i)
#pragma unroll
    for (int j = 0; j < 4; ++j)
#pragma unroll
      for (int e = 0; e < 8; ++e) {
        int gr = bm + wm + i * 16 + e + 8 * gh;
        int gc = bn + wn + j * 16 + col0;
        float acc = c[i][j][e] + bias[gc];
        if constexpr (EPI == 0) {
          outB[(size_t)gr * N + gc] = (__bf16)acc;
        } else if constexpr (EPI == 1) {
          outB[(size_t)gr * N + gc] = (__bf16)gelu_f(acc);
        } else {
          outF[(size_t)gr * N + gc] = acc + res[(size_t)gr * N + gc];
        }
      }
}

// ---------------------------------------------------------------------------
// Flash attention: grid (L/128, H, B), 8 waves x 16 query rows.
// KV in 32-key steps; K tile staged by TDM (padded rows), V staged transposed
// by the block. Online softmax with shuffle row-reductions; P routed through
// per-wave LDS to rebuild the A-fragment layout.
// ---------------------------------------------------------------------------
__global__ __launch_bounds__(256) void attn_kernel(const __bf16* __restrict__ qkv,
                                                   __bf16* __restrict__ y) {
  constexpr int LDKs = 72;  // Ks[key][d]: 64 elems (32 DWORDs) + 8 elems (4 DWORDs)
  constexpr int LDVs = 40;  // Vs[d][key] stride (transposed)
  constexpr int LDPs = 34;  // Pb[row][k] stride
  __shared__ __bf16 Ks[32 * LDKs];
  __shared__ __bf16 Vs[64 * LDVs];
  __shared__ __bf16 Pb[8 * 16 * LDPs];

  const int tid = threadIdx.x;
  const int w  = tid >> 5;
  const int l  = tid & 31;
  const int cl = l & 15;
  const int gh = l >> 4;
  const int b  = blockIdx.z;
  const int h  = blockIdx.y;
  const int qblk  = blockIdx.x * 128;
  const int qbase = qblk + w * 16;
  const size_t rstr = 3 * 1024;  // qkv row stride (3C)
  const __bf16* qptr = qkv + ((size_t)b * 2048) * rstr + h * 64;
  const __bf16* kptr = qptr + 1024;
  const __bf16* vptr = qptr + 2048;

  // Q fragments (A-layout), loaded once per wave straight from global.
  v16bf qa[2];
  {
    const __bf16* qp = qptr + (size_t)(qbase + cl) * rstr;
#pragma unroll
    for (int f = 0; f < 2; ++f)
#pragma unroll
      for (int v = 0; v < 8; ++v) {
        int kb = f * 32 + gh * 16 + v * 2;
        bf16x2 t = *(const bf16x2*)(qp + kb);
        qa[f][2 * v] = t.x; qa[f][2 * v + 1] = t.y;
      }
  }

  v8f o[4];
#pragma unroll
  for (int j = 0; j < 4; ++j) o[j] = v8f_zero();
  float m_run[8], l_run[8];
#pragma unroll
  for (int e = 0; e < 8; ++e) { m_run[e] = -1e30f; l_run[e] = 0.f; }

  const int kmax = qblk + 127;
  for (int kb = 0; kb <= kmax; kb += 32) {
    __syncthreads();
#if USE_TDM
    // K tile 32x64 via TDM: pad 32 DWORDs -> +4 DWORDs (stride 72 elems).
    if (w == 0) {
      tdm_load_2d_bf16((unsigned)(unsigned long long)(uintptr_t)Ks,
                       kptr + (size_t)kb * rstr,
                       /*tensor_d0=*/64, /*tensor_d1=*/32,
                       /*stride0=*/(unsigned long long)rstr,
                       /*tile_d0=*/64, /*tile_d1=*/32,
                       /*pad_interval=*/4, /*pad_amount=*/3);
    }
#endif
    {
      int e = tid * 8;
      int key = e >> 6, d = e & 63;
#if !USE_TDM
      bf16x8 kv = *(const bf16x8*)(kptr + (size_t)(kb + key) * rstr + d);
      *(bf16x8*)(Ks + (size_t)key * LDKs + d) = kv;
#endif
      bf16x8 vv = *(const bf16x8*)(vptr + (size_t)(kb + key) * rstr + d);
#pragma unroll
      for (int q2 = 0; q2 < 8; ++q2) Vs[(size_t)(d + q2) * LDVs + key] = vv[q2];
    }
#if USE_TDM
    if (w == 0) __builtin_amdgcn_s_wait_tensorcnt(0);
#endif
    __syncthreads();

    // S = Q K^T  (two 16x16 f32 tiles over 32 keys; K-dim = 64 via 2 WMMAs each)
    v8f s0 = v8f_zero(), s1 = v8f_zero();
#pragma unroll
    for (int f = 0; f < 2; ++f) {
      const __bf16* p0 = Ks + (size_t)cl * LDKs + f * 32 + gh * 16;
      const __bf16* p1 = Ks + (size_t)(16 + cl) * LDKs + f * 32 + gh * 16;
      v16bf kf0, kf1;
#pragma unroll
      for (int v = 0; v < 8; ++v) {
        bf16x2 t0 = *(const bf16x2*)(p0 + 2 * v);
        kf0[2 * v] = t0.x; kf0[2 * v + 1] = t0.y;
        bf16x2 t1 = *(const bf16x2*)(p1 + 2 * v);
        kf1[2 * v] = t1.x; kf1[2 * v + 1] = t1.y;
      }
      s0 = wmma_bf16(qa[f], kf0, s0);
      s1 = wmma_bf16(qa[f], kf1, s1);
    }

    // Online softmax update (rows live across 16 lanes of each half-group).
    const float scale = 0.125f;  // 1/sqrt(64)
#pragma unroll
    for (int e = 0; e < 8; ++e) {
      int qrow = qbase + e + 8 * gh;
      bool valid0 = (kb + cl) <= qrow;
      bool valid1 = (kb + 16 + cl) <= qrow;
      float v0 = s0[e] * scale, v1 = s1[e] * scale;
      float mx = fmaxf(valid0 ? v0 : -1e30f, valid1 ? v1 : -1e30f);
#pragma unroll
      for (int off = 1; off < 16; off <<= 1) mx = fmaxf(mx, __shfl_xor(mx, off));
      float mnew = fmaxf(m_run[e], mx);
      float fs = __expf(m_run[e] - mnew);
      float p0 = valid0 ? __expf(v0 - mnew) : 0.f;
      float p1 = valid1 ? __expf(v1 - mnew) : 0.f;
      float rsum = p0 + p1;
#pragma unroll
      for (int off = 1; off < 16; off <<= 1) rsum += __shfl_xor(rsum, off);
      l_run[e] = l_run[e] * fs + rsum;
      m_run[e] = mnew;
#pragma unroll
      for (int j = 0; j < 4; ++j) o[j][e] *= fs;
      __bf16* pb = Pb + (size_t)(w * 16 + e + 8 * gh) * LDPs;
      pb[cl]      = (__bf16)p0;
      pb[cl + 16] = (__bf16)p1;
    }

    // Re-gather P into A-layout (wave-private LDS region, per-wave ordering).
    v16bf pf;
    {
      const __bf16* pp = Pb + (size_t)(w * 16 + cl) * LDPs;
#pragma unroll
      for (int v = 0; v < 8; ++v) {
        int kk = gh * 16 + v * 2;
        bf16x2 t = *(const bf16x2*)(pp + kk);
        pf[2 * v] = t.x; pf[2 * v + 1] = t.y;
      }
    }
    // O += P @ V  (4 WMMAs over the 4 d-subtiles)
#pragma unroll
    for (int j = 0; j < 4; ++j) {
      const __bf16* vp = Vs + (size_t)(j * 16 + cl) * LDVs + gh * 16;
      v16bf vf;
#pragma unroll
      for (int v = 0; v < 8; ++v) {
        bf16x2 t = *(const bf16x2*)(vp + 2 * v);
        vf[2 * v] = t.x; vf[2 * v + 1] = t.y;
      }
      o[j] = wmma_bf16(pf, vf, o[j]);
    }
  }

  // Normalize and write y (bf16, [B,L,C] with head offset).
#pragma unroll
  for (int e = 0; e < 8; ++e) {
    float inv = 1.0f / l_run[e];
    int qrow = qbase + e + 8 * gh;
    __bf16* yp = y + ((size_t)b * 2048 + qrow) * 1024 + h * 64;
#pragma unroll
    for (int j = 0; j < 4; ++j) yp[j * 16 + cl] = (__bf16)(o[j][e] * inv);
  }
}

// ---------------------------------------------------------------------------
// Host launch: x -> LN1 -> QKV -> attn -> proj(+res) -> LN2 -> FC(gelu) -> FC2(+res)
// ---------------------------------------------------------------------------
extern "C" void kernel_launch(void* const* d_in, const int* in_sizes, int n_in,
                              void* d_out, int out_size, void* d_ws, size_t ws_size,
                              hipStream_t stream) {
  (void)in_sizes; (void)n_in; (void)out_size; (void)ws_size;
  const float* x     = (const float*)d_in[0];
  const float* ln1w  = (const float*)d_in[1];
  const float* ln1b  = (const float*)d_in[2];
  const float* wattn = (const float*)d_in[3];
  const float* battn = (const float*)d_in[4];
  const float* wproj = (const float*)d_in[5];
  const float* bproj = (const float*)d_in[6];
  const float* ln2w  = (const float*)d_in[7];
  const float* ln2b  = (const float*)d_in[8];
  const float* wfc   = (const float*)d_in[9];
  const float* bfc   = (const float*)d_in[10];
  const float* wfc2  = (const float*)d_in[11];
  const float* bfc2  = (const float*)d_in[12];

  const int M = 4096;  // B*L
  const int C = 1024;

  char* ws = (char*)d_ws;
  size_t off = 0;
  auto take = [&](size_t bytes) -> char* {
    char* p = ws + off;
    off += (bytes + 255) & ~(size_t)255;
    return p;
  };
  __bf16* wA   = (__bf16*)take((size_t)C * 3 * C * 2);
  __bf16* wP   = (__bf16*)take((size_t)C * C * 2);
  __bf16* wF   = (__bf16*)take((size_t)C * 4 * C * 2);
  __bf16* wF2  = (__bf16*)take((size_t)4 * C * C * 2);
  __bf16* hbuf = (__bf16*)take((size_t)M * C * 2);
  char* reg2   = take((size_t)M * 4 * C * 2);  // qkv+y, later reused as fc activation
  __bf16* qkvb  = (__bf16*)reg2;
  __bf16* ybuf  = (__bf16*)(reg2 + (size_t)M * 3 * C * 2);
  __bf16* fcact = (__bf16*)reg2;  // alias: qkv/y dead after proj GEMM
  float* x2 = (float*)take((size_t)M * C * 4);

  auto cvt = [&](const float* s, __bf16* d, size_t n) {
    int n4 = (int)(n / 4);
    cvt_kernel<<<(n4 + 255) / 256, 256, 0, stream>>>(s, d, n4);
  };
  cvt(wattn, wA,  (size_t)C * 3 * C);
  cvt(wproj, wP,  (size_t)C * C);
  cvt(wfc,   wF,  (size_t)C * 4 * C);
  cvt(wfc2,  wF2, (size_t)4 * C * C);

  // h = LN1(x)
  ln_kernel<<<M, 256, 0, stream>>>(x, ln1w, ln1b, hbuf);
  // qkv = h @ w_attn + b_attn  (bf16 out)
  gemm_kernel<0><<<dim3(3 * C / BN, M / BM), 256, 0, stream>>>(
      hbuf, wA, battn, nullptr, nullptr, qkvb, 3 * C, C);
  // y = attention(q,k,v)  (bf16 out)
  attn_kernel<<<dim3(2048 / 128, 16, 2), 256, 0, stream>>>(qkvb, ybuf);
  // x2 = y @ w_proj + b_proj + x  (f32 out)
  gemm_kernel<2><<<dim3(C / BN, M / BM), 256, 0, stream>>>(
      ybuf, wP, bproj, x, x2, nullptr, C, C);
  // h = LN2(x2)
  ln_kernel<<<M, 256, 0, stream>>>(x2, ln2w, ln2b, hbuf);
  // fcact = gelu(h @ w_fc + b_fc)  (bf16 out)
  gemm_kernel<1><<<dim3(4 * C / BN, M / BM), 256, 0, stream>>>(
      hbuf, wF, bfc, nullptr, nullptr, fcact, 4 * C, C);
  // out = fcact @ w_fc2 + b_fc2 + x2  (f32 out)
  gemm_kernel<2><<<dim3(C / BN, M / BM), 256, 0, stream>>>(
      fcact, wF2, bfc2, x2, (float*)d_out, nullptr, C, 4 * C);
}
